// NTXentLoss_41437844471874
// MI455X (gfx1250) — compile-verified, matrix operand order checked by reading, not run on hardware
//
#include <hip/hip_runtime.h>
#include <hip/hip_bf16.h>

typedef _Float16 v16h __attribute__((ext_vector_type(16)));
typedef _Float16 v8h  __attribute__((ext_vector_type(8)));
typedef _Float16 v4h  __attribute__((ext_vector_type(4)));
typedef float    v8f  __attribute__((ext_vector_type(8)));

#define NROWS 8192
#define DIM   128
#define TOT   (2 * NROWS)   // 16384
#define INV_T 2.0f          // 1 / 0.5

__device__ __forceinline__ float wave_sum32(float x) {
    x += __shfl_xor(x, 1, 32);
    x += __shfl_xor(x, 2, 32);
    x += __shfl_xor(x, 4, 32);
    x += __shfl_xor(x, 8, 32);
    x += __shfl_xor(x, 16, 32);
    return x;
}

// ---------------------------------------------------------------- kernel 0
__global__ void ntx_zero_kernel(float* __restrict__ denom, float* __restrict__ out) {
    int i = blockIdx.x * blockDim.x + threadIdx.x;
    if (i < TOT) denom[i] = 0.0f;
    if (i == 0) out[0] = 0.0f;
}

// ---------------------------------------------------------------- kernel 1
// One wave32 per row index i: normalize h1[i], h2[i], emit f16 rows into
// hf[i] and hf[NROWS+i], and the positive-pair dot.
__global__ void __launch_bounds__(256)
ntx_norm_kernel(const float* __restrict__ h1, const float* __restrict__ h2,
                _Float16* __restrict__ hf, float* __restrict__ pos_dot) {
    const int wave = threadIdx.x >> 5;
    const int lane = threadIdx.x & 31;
    const int row  = blockIdx.x * 8 + wave;   // grid sized exactly

    const float4* p1 = (const float4*)(h1 + (size_t)row * DIM);
    const float4* p2 = (const float4*)(h2 + (size_t)row * DIM);
    float4 a = p1[lane];
    float4 b = p2[lane];

    float s1 = a.x * a.x + a.y * a.y + a.z * a.z + a.w * a.w;
    float s2 = b.x * b.x + b.y * b.y + b.z * b.z + b.w * b.w;
    s1 = wave_sum32(s1);
    s2 = wave_sum32(s2);
    const float r1 = 1.0f / fmaxf(sqrtf(s1), 1e-12f);
    const float r2 = 1.0f / fmaxf(sqrtf(s2), 1e-12f);

    a.x *= r1; a.y *= r1; a.z *= r1; a.w *= r1;
    b.x *= r2; b.y *= r2; b.z *= r2; b.w *= r2;

    float d = a.x * b.x + a.y * b.y + a.z * b.z + a.w * b.w;
    d = wave_sum32(d);

    v4h ha = { (_Float16)a.x, (_Float16)a.y, (_Float16)a.z, (_Float16)a.w };
    v4h hb = { (_Float16)b.x, (_Float16)b.y, (_Float16)b.z, (_Float16)b.w };
    *(v4h*)(hf + (size_t)row * DIM + lane * 4)            = ha;
    *(v4h*)(hf + (size_t)(NROWS + row) * DIM + lane * 4)  = hb;

    if (lane == 0) pos_dot[row] = d;
}

// ---------------------------------------------------------------- kernel 2
// Per-lane fragment load matching the 16-bit 16x32 A layout:
// lanes 0-15 hold K 0-7 / 16-23, lanes 16-31 hold K 8-15 / 24-31.
// The B (32x16) fragment for H*H^T uses the identical pattern with the lane
// indexing the N column's source row — no transpose load needed.
// All row offsets fit in 32 bits; precompute per-lane byte offsets once so
// the unrolled K loop becomes base + uint + constant-immediate loads.
__device__ __forceinline__ v16h ntx_frag(const char* __restrict__ base,
                                         unsigned byteoff, int k0) {
    v8h lo = *(const v8h*)(base + byteoff + k0 * 2);
    v8h hh = *(const v8h*)(base + byteoff + k0 * 2 + 32);
    return __builtin_shufflevector(lo, hh, 0, 1, 2, 3, 4, 5, 6, 7,
                                          8, 9, 10, 11, 12, 13, 14, 15);
}

__global__ void __launch_bounds__(256)
ntx_gram_kernel(const _Float16* __restrict__ hf, float* __restrict__ denom) {
    const int lane = threadIdx.x & 31;
    const int wave = threadIdx.x >> 5;
    const int m    = lane & 15;
    const int hi   = lane >> 4;

    const int rowBase = blockIdx.y * 128 + wave * 16;  // 16-row strip per wave
    const int colBase = blockIdx.x * 128;              // 128-col strip per block

    const char* base = (const char*)hf;

    // per-lane byte offsets (include the hi-half K skew of 8 halves = 16 B)
    const unsigned aoff = (unsigned)(((rowBase + m) * DIM + hi * 8) * 2);
    unsigned boff[8];
#pragma unroll
    for (int nt = 0; nt < 8; ++nt)
        boff[nt] = (unsigned)(((colBase + nt * 16 + m) * DIM + hi * 8) * 2);

    const v8f vzero = {0.f, 0.f, 0.f, 0.f, 0.f, 0.f, 0.f, 0.f};
    v8f acc[8];
#pragma unroll
    for (int i = 0; i < 8; ++i) acc[i] = vzero;

#pragma unroll
    for (int k0 = 0; k0 < DIM; k0 += 32) {
        const v16h afrag = ntx_frag(base, aoff, k0);
#pragma unroll
        for (int nt = 0; nt < 8; ++nt) {
            const v16h bfrag = ntx_frag(base, boff[nt], k0);
            acc[nt] = __builtin_amdgcn_wmma_f32_16x16x32_f16(
                false, afrag, false, bfrag, (short)0, acc[nt], false, false);
        }
    }

    // epilogue: exp(dot / T) and row-sum over the 128-column strip
    v8f tot = vzero;
#pragma unroll
    for (int nt = 0; nt < 8; ++nt) {
#pragma unroll
        for (int e = 0; e < 8; ++e)
            tot[e] += __expf(acc[nt][e] * INV_T);
    }

    // C/D layout: VGPR e, lanes 0-15 -> (M=e, N=lane); lanes 16-31 -> (M=e+8).
    // xor masks 1,2,4,8 reduce within each 16-lane half (halves stay separate).
#pragma unroll
    for (int e = 0; e < 8; ++e) {
        float s = tot[e];
        s += __shfl_xor(s, 1, 32);
        s += __shfl_xor(s, 2, 32);
        s += __shfl_xor(s, 4, 32);
        s += __shfl_xor(s, 8, 32);
        if (m == 0) atomicAdd(&denom[rowBase + hi * 8 + e], s);
    }
}

// ---------------------------------------------------------------- kernel 3
__global__ void __launch_bounds__(256)
ntx_final_kernel(const _Float16* __restrict__ hf, const float* __restrict__ denom,
                 const float* __restrict__ pos_dot, float* __restrict__ out) {
    __shared__ float sbuf[256];
    const int i = blockIdx.x * blockDim.x + threadIdx.x;  // 0..16383

    // diagonal term, recomputed in f32 from the same f16 data the GEMM used
    const _Float16* row = hf + (size_t)i * DIM;
    float d = 0.0f;
#pragma unroll
    for (int k = 0; k < DIM; k += 4) {
        v4h v = *(const v4h*)(row + k);
        float x0 = (float)v[0], x1 = (float)v[1], x2 = (float)v[2], x3 = (float)v[3];
        d += x0 * x0 + x1 * x1 + x2 * x2 + x3 * x3;
    }
    const float diag = __expf(d * INV_T);
    const float pd   = pos_dot[(i < NROWS) ? i : (i - NROWS)];
    // -log(pos/denom) = log(denom) - dot/T
    const float li = __logf(denom[i] - diag) - pd * INV_T;

    sbuf[threadIdx.x] = li;
    __syncthreads();
#pragma unroll
    for (int s = 128; s > 0; s >>= 1) {
        if (threadIdx.x < s) sbuf[threadIdx.x] += sbuf[threadIdx.x + s];
        __syncthreads();
    }
    if (threadIdx.x == 0) atomicAdd(out, sbuf[0] * (1.0f / (float)TOT));
}

// ---------------------------------------------------------------- launch
extern "C" void kernel_launch(void* const* d_in, const int* in_sizes, int n_in,
                              void* d_out, int out_size, void* d_ws, size_t ws_size,
                              hipStream_t stream) {
    const float* h1 = (const float*)d_in[0];
    const float* h2 = (const float*)d_in[1];
    float* out = (float*)d_out;

    char* ws = (char*)d_ws;
    _Float16* hf   = (_Float16*)ws;                               // 16384*128*2 = 4 MB
    size_t off     = (size_t)TOT * DIM * sizeof(_Float16);
    float* denom   = (float*)(ws + off);                          // 64 KB
    off           += (size_t)TOT * sizeof(float);
    float* pos_dot = (float*)(ws + off);                          // 32 KB

    ntx_zero_kernel<<<(TOT + 255) / 256, 256, 0, stream>>>(denom, out);
    ntx_norm_kernel<<<NROWS / 8, 256, 0, stream>>>(h1, h2, hf, pos_dot);

    dim3 grid(TOT / 128, TOT / 128);   // 128 x 128 blocks of 128x128 tiles
    ntx_gram_kernel<<<grid, 256, 0, stream>>>(hf, denom);

    ntx_final_kernel<<<TOT / 256, 256, 0, stream>>>(hf, denom, pos_dot, out);
}
